// Backbone_36189394436309
// MI455X (gfx1250) — matrix operand order repaired and checked by reading, not compiled, not run on hardware
//
#include <hip/hip_runtime.h>
#include <hip/hip_bf16.h>

typedef _Float16 v16h __attribute__((ext_vector_type(16)));
typedef float    v8f  __attribute__((ext_vector_type(8)));

#if __has_builtin(__builtin_amdgcn_tanhf)
#define TANHF(x) __builtin_amdgcn_tanhf(x)
#else
#define TANHF(x) tanhf(x)
#endif

// ---------------- constants ----------------
#define OBS_DIM   512
#define HID       128
#define W1E       (128*512)        // 65536 f16 elems in ws
#define W2OFF     W1E              // 65536
#define W3OFF     (W1E + 128*128)  // 81920
#define TOTW      (W3OFF + 128*128)// 98304
#define PW1       264              // LDS pitch (halves) for staged w1 K-half (256+8)
#define PW2       136              // LDS pitch (halves) for w2/w3 (128+8)
#define PX        136              // LDS pitch (halves) for activation tile
#define WREG_H    34816            // halves in weight region: 2*128*136 (>= 128*264)
#define XREG_H    (8*16*PX)        // 17408 halves
#define SH_BYTES  ((WREG_H + XREG_H) * 2)  // 104448 bytes dynamic LDS

// ---------------- fragment helpers ----------------
static __device__ __forceinline__ v8f wmma16(v16h a, v16h b, v8f c) {
  // (neg_a, A, neg_b, B, c_mod, C, reuse_a, reuse_b)
  return __builtin_amdgcn_wmma_f32_16x16x32_f16(false, a, false, b, (short)0, c,
                                                false, false);
}

// B fragment per lane: elems 0..7 = p[0..7], 8..15 = p[16..23]: two b128 loads
static __device__ __forceinline__ v16h load16h(const _Float16* p) {
  union { uint4 u[2]; v16h h; } t;
  t.u[0] = ((const uint4*)p)[0];
  t.u[1] = ((const uint4*)p)[2];
  return t.h;
}

// A fragment from an f16 row: elems 0..7 = row[k0..k0+7], 8..15 = row[k0+16..k0+23]
static __device__ __forceinline__ v16h load_a_f16(const _Float16* row, int k0) {
  union { uint4 u[2]; v16h h; } t;
  t.u[0] = *(const uint4*)(row + k0);
  t.u[1] = *(const uint4*)(row + k0 + 16);
  return t.h;
}

// A fragment from an fp32 row (obs), converted to f16 in registers
static __device__ __forceinline__ v16h load_a_f32(const float* p) {
  float4 c0 = ((const float4*)p)[0];
  float4 c1 = ((const float4*)p)[1];
  float4 c2 = ((const float4*)p)[4];   // p + 16
  float4 c3 = ((const float4*)p)[5];   // p + 20
  v16h a;
  a[0]=(_Float16)c0.x; a[1]=(_Float16)c0.y; a[2]=(_Float16)c0.z; a[3]=(_Float16)c0.w;
  a[4]=(_Float16)c1.x; a[5]=(_Float16)c1.y; a[6]=(_Float16)c1.z; a[7]=(_Float16)c1.w;
  a[8]=(_Float16)c2.x; a[9]=(_Float16)c2.y; a[10]=(_Float16)c2.z; a[11]=(_Float16)c2.w;
  a[12]=(_Float16)c3.x; a[13]=(_Float16)c3.y; a[14]=(_Float16)c3.z; a[15]=(_Float16)c3.w;
  return a;
}

// ---------------- weight conversion pre-kernel ----------------
__global__ void cvt_weights(const float* __restrict__ w1,
                            const float* __restrict__ w2,
                            const float* __restrict__ w3,
                            _Float16* __restrict__ ws) {
  int i = blockIdx.x * blockDim.x + threadIdx.x;
  if (i < W1E)                 ws[i] = (_Float16)w1[i];
  else if (i < W3OFF)          ws[i] = (_Float16)w2[i - W2OFF];
  else if (i < TOTW)           ws[i] = (_Float16)w3[i - W3OFF];
}

// ---------------- register-resident bias + LN + tanh ----------------
static __device__ __forceinline__ void bias_ln_tanh(
    v8f acc[8], const float* __restrict__ bias,
    const float* __restrict__ g, const float* __restrict__ be,
    int ln, bool do_ln) {
#pragma unroll
  for (int t = 0; t < 8; ++t) {
    float bt = bias[t * 16 + ln];
#pragma unroll
    for (int v = 0; v < 8; ++v) acc[t][v] += bt;
  }
  if (do_ln) {
    float s[8], q[8];
#pragma unroll
    for (int v = 0; v < 8; ++v) { s[v] = 0.f; q[v] = 0.f; }
#pragma unroll
    for (int t = 0; t < 8; ++t)
#pragma unroll
      for (int v = 0; v < 8; ++v) { s[v] += acc[t][v]; q[v] += acc[t][v] * acc[t][v]; }
    // reduce over the 16 lanes of this half (xor masks stay within the half)
#pragma unroll
    for (int off = 1; off <= 8; off <<= 1) {
#pragma unroll
      for (int v = 0; v < 8; ++v) {
        s[v] += __shfl_xor(s[v], off, 32);
        q[v] += __shfl_xor(q[v], off, 32);
      }
    }
    float mu[8], rstd[8];
#pragma unroll
    for (int v = 0; v < 8; ++v) {
      mu[v] = s[v] * (1.0f / 128.0f);
      float var = q[v] * (1.0f / 128.0f) - mu[v] * mu[v];
      rstd[v] = rsqrtf(var + 1e-5f);
    }
#pragma unroll
    for (int t = 0; t < 8; ++t) {
      float gg = g[t * 16 + ln];
      float bb = be[t * 16 + ln];
#pragma unroll
      for (int v = 0; v < 8; ++v)
        acc[t][v] = TANHF((acc[t][v] - mu[v]) * rstd[v] * gg + bb);
    }
  } else {
#pragma unroll
    for (int t = 0; t < 8; ++t)
#pragma unroll
      for (int v = 0; v < 8; ++v) acc[t][v] = TANHF(acc[t][v]);
  }
}

// ---------------- fused MLP kernel ----------------
__global__ void __launch_bounds__(256, 2)
mlp_fused(const float* __restrict__ obs, const _Float16* __restrict__ ws,
          const float* __restrict__ b1, const float* __restrict__ g1,
          const float* __restrict__ be1,
          const float* __restrict__ b2, const float* __restrict__ g2,
          const float* __restrict__ be2,
          const float* __restrict__ b3, float* __restrict__ out) {
  extern __shared__ __align__(16) unsigned char smem_raw[];
  _Float16* wbuf = (_Float16*)smem_raw;          // WREG_H halves
  _Float16* xbuf = wbuf + WREG_H;                // XREG_H halves

  const int tid  = threadIdx.x;
  const int lane = tid & 31;
  const int wv   = tid >> 5;
  const int hh   = lane >> 4;     // 0: lanes 0-15, 1: lanes 16-31
  const int ln   = lane & 15;
  const size_t rowBase = (size_t)blockIdx.x * 128 + (size_t)wv * 16;
  const float* myrow = obs + (rowBase + (size_t)ln) * OBS_DIM;

  // ---- stage w1 K-half 0 into LDS (rows of 256 halves, pitch PW1) ----
#pragma unroll
  for (int i = 0; i < 16; ++i) {
    int c = i * 256 + tid;            // 4096 chunks of 8 halves
    int r = c >> 5, cc = c & 31;
    *(uint4*)(wbuf + r * PW1 + cc * 8) = *(const uint4*)(ws + r * OBS_DIM + cc * 8);
  }
  __syncthreads();

  v8f acc[8];
#pragma unroll
  for (int t = 0; t < 8; ++t) acc[t] = v8f{0.f,0.f,0.f,0.f,0.f,0.f,0.f,0.f};

  // ---- GEMM1, K = 0..255 ----
#pragma unroll
  for (int kb = 0; kb < 256; kb += 32) {
    __builtin_prefetch(myrow + kb + 256, 0, 1);
    v16h a = load_a_f32(myrow + kb + hh * 8);
#pragma unroll
    for (int t = 0; t < 8; ++t)
      acc[t] = wmma16(a, load16h(wbuf + (t * 16 + ln) * PW1 + kb + hh * 8), acc[t]);
  }
  __syncthreads();

  // ---- stage w1 K-half 1 ----
#pragma unroll
  for (int i = 0; i < 16; ++i) {
    int c = i * 256 + tid;
    int r = c >> 5, cc = c & 31;
    *(uint4*)(wbuf + r * PW1 + cc * 8) =
        *(const uint4*)(ws + r * OBS_DIM + 256 + cc * 8);
  }
  __syncthreads();

  // ---- GEMM1, K = 256..511 ----
#pragma unroll
  for (int kb = 256; kb < 512; kb += 32) {
    v16h a = load_a_f32(myrow + kb + hh * 8);
#pragma unroll
    for (int t = 0; t < 8; ++t)
      acc[t] = wmma16(a, load16h(wbuf + (t * 16 + ln) * PW1 + (kb - 256) + hh * 8), acc[t]);
  }
  __syncthreads();

  // ---- stage w2 + w3 (pitch PW2 each, w3 at +128*PW2) ----
#pragma unroll
  for (int i = 0; i < 16; ++i) {
    int c = i * 256 + tid;            // 4096 chunks of 8 halves
    if (c < 2048) {
      int r = c >> 4, cc = c & 15;
      *(uint4*)(wbuf + r * PW2 + cc * 8) =
          *(const uint4*)(ws + W2OFF + r * HID + cc * 8);
    } else {
      int c2 = c - 2048;
      int r = c2 >> 4, cc = c2 & 15;
      *(uint4*)(wbuf + 128 * PW2 + r * PW2 + cc * 8) =
          *(const uint4*)(ws + W3OFF + r * HID + cc * 8);
    }
  }

  // ---- layer1 epilogue: bias + LN + tanh (registers), stash f16 tile ----
  bias_ln_tanh(acc, b1, g1, be1, ln, true);
  _Float16* xrow = xbuf + wv * 16 * PX;
#pragma unroll
  for (int t = 0; t < 8; ++t)
#pragma unroll
    for (int v = 0; v < 8; ++v)
      xrow[(hh * 8 + v) * PX + t * 16 + ln] = (_Float16)acc[t][v];
  __syncthreads();   // also guarantees w2/w3 staging is visible

  // ---- GEMM2, K = 128 ----
  const _Float16* arow = xbuf + (wv * 16 + ln) * PX;
#pragma unroll
  for (int t = 0; t < 8; ++t) acc[t] = v8f{0.f,0.f,0.f,0.f,0.f,0.f,0.f,0.f};
#pragma unroll
  for (int kb = 0; kb < 128; kb += 32) {
    v16h a = load_a_f16(arow, kb + hh * 8);
#pragma unroll
    for (int t = 0; t < 8; ++t)
      acc[t] = wmma16(a, load16h(wbuf + (t * 16 + ln) * PW2 + kb + hh * 8), acc[t]);
  }

  // ---- layer2 epilogue (per-wave private tile: program order suffices) ----
  bias_ln_tanh(acc, b2, g2, be2, ln, true);
#pragma unroll
  for (int t = 0; t < 8; ++t)
#pragma unroll
    for (int v = 0; v < 8; ++v)
      xrow[(hh * 8 + v) * PX + t * 16 + ln] = (_Float16)acc[t][v];

  // ---- GEMM3, K = 128 ----
  const _Float16* w3l = wbuf + 128 * PW2;
#pragma unroll
  for (int t = 0; t < 8; ++t) acc[t] = v8f{0.f,0.f,0.f,0.f,0.f,0.f,0.f,0.f};
#pragma unroll
  for (int kb = 0; kb < 128; kb += 32) {
    v16h a = load_a_f16(arow, kb + hh * 8);
#pragma unroll
    for (int t = 0; t < 8; ++t)
      acc[t] = wmma16(a, load16h(w3l + (t * 16 + ln) * PW2 + kb + hh * 8), acc[t]);
  }

  // ---- layer3 epilogue: bias + tanh, store fp32 output ----
  bias_ln_tanh(acc, b3, b3, b3, ln, false);
#pragma unroll
  for (int t = 0; t < 8; ++t)
#pragma unroll
    for (int v = 0; v < 8; ++v)
      out[((rowBase + (size_t)(hh * 8 + v)) << 7) + t * 16 + ln] = acc[t][v];
}

// ---------------- host launcher ----------------
extern "C" void kernel_launch(void* const* d_in, const int* in_sizes, int n_in,
                              void* d_out, int out_size, void* d_ws, size_t ws_size,
                              hipStream_t stream) {
  const float* obs = (const float*)d_in[0];
  const float* w1  = (const float*)d_in[1];
  const float* b1  = (const float*)d_in[2];
  const float* g1  = (const float*)d_in[3];
  const float* be1 = (const float*)d_in[4];
  const float* w2  = (const float*)d_in[5];
  const float* b2  = (const float*)d_in[6];
  const float* g2  = (const float*)d_in[7];
  const float* be2 = (const float*)d_in[8];
  const float* w3  = (const float*)d_in[9];
  const float* b3  = (const float*)d_in[10];
  float* out = (float*)d_out;
  _Float16* ws = (_Float16*)d_ws;

  const int B = in_sizes[0] / OBS_DIM;   // 262144

  cvt_weights<<<(TOTW + 255) / 256, 256, 0, stream>>>(w1, w2, w3, ws);
  mlp_fused<<<B / 128, 256, SH_BYTES, stream>>>(obs, ws, b1, g1, be1,
                                                b2, g2, be2, b3, out);
}
